// Attention_60576218743412
// MI455X (gfx1250) — matrix operand order repaired and checked by reading, not compiled, not run on hardware
//
#include <hip/hip_runtime.h>
#include <hip/hip_bf16.h>

#define DIM    768
#define HEADS  12
#define DHEAD  64
#define SEQ    2048
#define BATCH  4
#define NTOK   (BATCH*SEQ)   /* 8192 */
#define EQKV   (3*DIM)       /* 2304 */

typedef __attribute__((ext_vector_type(16))) __bf16 v16bf;
typedef __attribute__((ext_vector_type(8)))  __bf16 bf16x8;
typedef __attribute__((ext_vector_type(8)))  float  v8f;

union Frag { v16bf v; bf16x8 h[2]; };

__device__ __forceinline__ v8f wmma_bf16(v16bf a, v16bf b, v8f c) {
  // D = A(16x32) * B(32x16) + C, f32 accumulate
  return __builtin_amdgcn_wmma_f32_16x16x32_bf16(false, a, false, b, (short)0, c, false, false);
}

// A-operand fragment (sequential K per lane-half): lane holds K = hi*16 .. hi*16+15
__device__ __forceinline__ v16bf load_a_frag(const __bf16* p, int hi) {
  Frag f;
  f.h[0] = *reinterpret_cast<const bf16x8*>(p + hi * 16);
  f.h[1] = *reinterpret_cast<const bf16x8*>(p + hi * 16 + 8);
  return f.v;
}
// B-operand fragment (sequential K per lane-half): lane holds K = hi*16 .. hi*16+15
__device__ __forceinline__ v16bf load_b_frag(const __bf16* p, int hi) {
  Frag f;
  f.h[0] = *reinterpret_cast<const bf16x8*>(p + hi * 16);
  f.h[1] = *reinterpret_cast<const bf16x8*>(p + hi * 16 + 8);
  return f.v;
}

// gfx1250 async copy: global -> LDS, 16B per lane, tracked by ASYNCcnt
__device__ __forceinline__ void async_copy_b128(unsigned lds_addr, unsigned long long gaddr) {
  asm volatile("global_load_async_to_lds_b128 %0, %1, off"
               :: "v"(lds_addr), "v"(gaddr) : "memory");
}

// ---------------------------------------------------------------- LN + cast
__global__ __launch_bounds__(256) void ln_cast_kernel(
    const float* __restrict__ x, const float* __restrict__ g,
    const float* __restrict__ bta, __bf16* __restrict__ xn) {
  const int row = blockIdx.x;
  const int tid = threadIdx.x;
  const float* xr = x + (size_t)row * DIM;
  float v0 = xr[tid], v1 = xr[tid + 256], v2 = xr[tid + 512];
  float s  = v0 + v1 + v2;
  float ss = v0 * v0 + v1 * v1 + v2 * v2;
  #pragma unroll
  for (int off = 16; off > 0; off >>= 1) {
    s  += __shfl_down(s,  off);
    ss += __shfl_down(ss, off);
  }
  __shared__ float sh[16];
  const int lane = tid & 31, wv = tid >> 5;
  if (lane == 0) { sh[wv] = s; sh[8 + wv] = ss; }
  __syncthreads();
  float tot = 0.f, tot2 = 0.f;
  #pragma unroll
  for (int w = 0; w < 8; ++w) { tot += sh[w]; tot2 += sh[8 + w]; }
  const float mu  = tot * (1.0f / DIM);
  const float var = tot2 * (1.0f / DIM) - mu * mu;
  const float rs  = rsqrtf(var + 1e-5f);
  __bf16* xo = xn + (size_t)row * DIM;
  xo[tid]       = (__bf16)((v0 - mu) * rs * g[tid]       + bta[tid]);
  xo[tid + 256] = (__bf16)((v1 - mu) * rs * g[tid + 256] + bta[tid + 256]);
  xo[tid + 512] = (__bf16)((v2 - mu) * rs * g[tid + 512] + bta[tid + 512]);
}

__global__ __launch_bounds__(256) void cast_bf16_kernel(
    const float* __restrict__ in, __bf16* __restrict__ out, int n) {
  int i = blockIdx.x * 256 + threadIdx.x;
  if (i < n) out[i] = (__bf16)in[i];
}

// ------------------------------------------------- generic bf16 WMMA GEMM
// C[m, e] = sum_d A[m, d] * W[e, d]   (A: MxK row-major, W: NxK row-major)
// Wave tile: 64(M) x 64(N) -> 16 accumulators.  Block = 8 waves -> 512(M).
template <typename OutT>
__global__ __launch_bounds__(256) void gemm_bf16_nt(
    const __bf16* __restrict__ A, const __bf16* __restrict__ W,
    OutT* __restrict__ C, int M, int Ncols, int K) {
  const int lane = threadIdx.x & 31, wv = threadIdx.x >> 5;
  const int lr = lane & 15, hi = lane >> 4;
  const int row0 = blockIdx.y * 512 + wv * 64;
  const int col0 = blockIdx.x * 64;
  const __bf16* ap[4];
  const __bf16* wp[4];
  #pragma unroll
  for (int i = 0; i < 4; ++i) {
    ap[i] = A + (size_t)(row0 + i * 16 + lr) * K;
    wp[i] = W + (size_t)(col0 + i * 16 + lr) * K;
  }
  v8f acc[4][4] = {};
  for (int kb = 0; kb < K; kb += 32) {
    v16bf a[4];
    #pragma unroll
    for (int i = 0; i < 4; ++i) a[i] = load_a_frag(ap[i] + kb, hi);
    #pragma unroll
    for (int j = 0; j < 4; ++j) {
      v16bf bw = load_b_frag(wp[j] + kb, hi);
      #pragma unroll
      for (int i = 0; i < 4; ++i) acc[i][j] = wmma_bf16(a[i], bw, acc[i][j]);
    }
  }
  #pragma unroll
  for (int i = 0; i < 4; ++i)
    #pragma unroll
    for (int j = 0; j < 4; ++j)
      #pragma unroll
      for (int v = 0; v < 8; ++v) {
        int row = row0 + i * 16 + 8 * hi + v;   // C/D layout: M = v + 8*hi
        int col = col0 + j * 16 + lr;           // N = lane & 15
        C[(size_t)row * Ncols + col] = (OutT)acc[i][j][v];
      }
}

// ------------------------------------------- RoPE + head split + V transpose
// qkv: [NTOK, 2304] bf16 -> Qr (pre-scaled by 1/8), Kr: [b,h,n,64]; Vt: [b,h,64,n]
__global__ __launch_bounds__(96) void rope_split_kernel(
    const __bf16* __restrict__ qkv, const float* __restrict__ coords,
    __bf16* __restrict__ Qr, __bf16* __restrict__ Kr, __bf16* __restrict__ Vt) {
  const int idx = blockIdx.x;             // b*SEQ*HEADS + n*HEADS + h
  const int b = idx / (SEQ * HEADS);
  const int rem = idx % (SEQ * HEADS);
  const int n = rem / HEADS;
  const int h = rem % HEADS;
  const int tid = threadIdx.x;
  const __bf16* row = qkv + ((size_t)b * SEQ + n) * EQKV;
  const size_t bh = (size_t)(b * HEADS + h);
  if (tid < 64) {
    const int unit = tid >> 5;            // 0 = q, 1 = k
    const int t = tid & 31;
    const int axis = t >> 4;              // axis 0 -> dims [0,32), axis 1 -> [32,64)
    const int i = t & 15;
    const float coord = coords[((size_t)b * SEQ + n) * 2 + axis];
    const float inv_freq = __expf(-((float)i * (1.0f / 16.0f)) * 9.0109133472f); // ln(8192)
    const float f = coord * inv_freq;
    float sn, cs;
    __sincosf(f, &sn, &cs);
    const float sc = unit ? 1.0f : 0.125f;  // fold DIM_HEAD^-0.5 into Q
    const int e0 = unit * DIM + h * DHEAD + axis * 32;
    const float t1 = (float)row[e0 + 2 * i];
    const float t2 = (float)row[e0 + 2 * i + 1];
    __bf16* dst = (unit ? Kr : Qr) + (bh * SEQ + n) * DHEAD + axis * 32;
    dst[i]      = (__bf16)((t1 * cs - t2 * sn) * sc);
    dst[16 + i] = (__bf16)((t1 * sn + t2 * cs) * sc);
  } else {
    const int t = tid - 64;               // 32 threads, 2 dims each
    #pragma unroll
    for (int u = 0; u < 2; ++u) {
      const int d = 2 * t + u;
      Vt[(bh * DHEAD + d) * SEQ + n] = row[2 * DIM + h * DHEAD + d];
    }
  }
}

// --------------------------------------------------- flash attention (WMMA)
// Per wave: 32 query rows (2 fragment sets). KV tiles of 32 staged in LDS via
// async global->LDS DMA, double-buffered on ASYNCcnt. S^T = K*Q^T keeps the
// softmax reduction in-lane; O^T += V^T * P^T.
#define KV_TILE 32
#define KROWB   144                 /* K tile LDS row pitch (64*2 + 16 pad) */
#define VROWB   80                  /* V tile LDS row pitch (32*2 + 16 pad) */
#define KBYTES  (32 * KROWB)        /* 4608 */
#define VBYTES  (64 * VROWB)        /* 5120 */
#define TILEB   (KBYTES + VBYTES)   /* 9728 */

__global__ __launch_bounds__(256) void attn_kernel(
    const __bf16* __restrict__ Q, const __bf16* __restrict__ Km,
    const __bf16* __restrict__ Vt, __bf16* __restrict__ O) {
  __shared__ __attribute__((aligned(16))) char smem[2][TILEB];
  const int bh = blockIdx.x;
  const int b = bh / HEADS, h = bh % HEADS;
  const int t = threadIdx.x;
  const int lane = t & 31, wv = t >> 5;
  const int lr = lane & 15, hi = lane >> 4;
  const int q0 = blockIdx.y * 256 + wv * 32;

  const __bf16* kbase = Km + (size_t)bh * SEQ * DHEAD;
  const __bf16* vbase = Vt + (size_t)bh * DHEAD * SEQ;

  // one K chunk (32x8 of 16B) + one V chunk (64x4 of 16B) per thread per tile
  const int kr = t >> 3, kc = t & 7;
  const int vr = t >> 2, vc = t & 3;
  auto stage = [&](int it, int buf) {
    const int kv = it * KV_TILE;
    const unsigned lbase = (unsigned)(uintptr_t)&smem[buf][0];
    async_copy_b128(lbase + kr * KROWB + kc * 16,
        (unsigned long long)(uintptr_t)(kbase + (size_t)(kv + kr) * DHEAD) + kc * 16);
    async_copy_b128(lbase + KBYTES + vr * VROWB + vc * 16,
        (unsigned long long)(uintptr_t)(vbase + (size_t)vr * SEQ + kv) + vc * 16);
  };

  // Q^T B-fragments for both 16-row query sets (Q pre-scaled by 1/8)
  v16bf qb[2][2];
  #pragma unroll
  for (int s = 0; s < 2; ++s) {
    const __bf16* qrow = Q + ((size_t)bh * SEQ + q0 + s * 16 + lr) * DHEAD;
    qb[s][0] = load_b_frag(qrow, hi);
    qb[s][1] = load_b_frag(qrow + 32, hi);
  }

  v8f o[2][4] = {};
  float m_run[2] = {-3.0e38f, -3.0e38f};
  float l_run[2] = {0.f, 0.f};

  stage(0, 0);
  stage(1, 1);
  const int NT = SEQ / KV_TILE;   // 64
  for (int it = 0; it < NT; ++it) {
    const int buf = it & 1;
    if (it + 1 < NT) asm volatile("s_wait_asynccnt 0x2" ::: "memory");
    else             asm volatile("s_wait_asynccnt 0x0" ::: "memory");
    __syncthreads();

    const __bf16* kt = (const __bf16*)&smem[buf][0];
    const __bf16* vt = (const __bf16*)&smem[buf][KBYTES];
    v16bf ka[2][2];
    #pragma unroll
    for (int mh = 0; mh < 2; ++mh)
      #pragma unroll
      for (int ks = 0; ks < 2; ++ks)
        ka[mh][ks] = load_a_frag(kt + (mh * 16 + lr) * (KROWB / 2) + ks * 32, hi);

    Frag pb[2];
    float alpha[2];
    #pragma unroll
    for (int s = 0; s < 2; ++s) {
      v8f st0 = {};                               // S^T rows kv .. kv+15
      st0 = wmma_bf16(ka[0][0], qb[s][0], st0);
      st0 = wmma_bf16(ka[0][1], qb[s][1], st0);
      v8f st1 = {};                               // S^T rows kv+16 .. kv+31
      st1 = wmma_bf16(ka[1][0], qb[s][0], st1);
      st1 = wmma_bf16(ka[1][1], qb[s][1], st1);

      float mx = -3.0e38f;
      #pragma unroll
      for (int e = 0; e < 8; ++e) mx = fmaxf(mx, fmaxf(st0[e], st1[e]));
      mx = fmaxf(mx, __shfl_xor(mx, 16));         // other kv half of this query
      const float m_new = fmaxf(m_run[s], mx);
      alpha[s] = __expf(m_run[s] - m_new);
      float ls = 0.f;
      #pragma unroll
      for (int e = 0; e < 8; ++e) {
        const float p0 = __expf(st0[e] - m_new);
        const float p1 = __expf(st1[e] - m_new);
        ls += p0 + p1;
        // B layout wants K = hi*16+e sequential; swap halves across lane pairs.
        const float x0 = __shfl_xor(p0, 16);
        const float x1 = __shfl_xor(p1, 16);
        pb[s].h[0][e] = (__bf16)(hi ? x1 : p0);   // K = hi*16 + e
        pb[s].h[1][e] = (__bf16)(hi ? p1 : x0);   // K = hi*16 + 8 + e
      }
      l_run[s] = l_run[s] * alpha[s] + ls + __shfl_xor(ls, 16);
      m_run[s] = m_new;
    }

    #pragma unroll
    for (int j = 0; j < 4; ++j) {
      v16bf va = load_a_frag(vt + (j * 16 + lr) * (VROWB / 2), hi);
      #pragma unroll
      for (int s = 0; s < 2; ++s) {
        #pragma unroll
        for (int e = 0; e < 8; ++e) o[s][j][e] *= alpha[s];
        o[s][j] = wmma_bf16(va, pb[s].v, o[s][j]);   // O^T += V^T * P^T
      }
    }

    __syncthreads();
    if (it + 2 < NT) stage(it + 2, buf);
  }

  #pragma unroll
  for (int s = 0; s < 2; ++s) {
    const float inv = 1.0f / l_run[s];
    __bf16* orow = O + ((size_t)b * SEQ + q0 + s * 16 + lr) * (HEADS * DHEAD) + h * DHEAD;
    #pragma unroll
    for (int j = 0; j < 4; ++j)
      #pragma unroll
      for (int v = 0; v < 8; ++v)
        orow[j * 16 + 8 * hi + v] = (__bf16)(o[s][j][v] * inv);
  }
}

// ---------------------------------------------------------------- launcher
extern "C" void kernel_launch(void* const* d_in, const int* in_sizes, int n_in,
                              void* d_out, int out_size, void* d_ws, size_t ws_size,
                              hipStream_t stream) {
  (void)in_sizes; (void)n_in; (void)out_size; (void)ws_size;
  const float* x      = (const float*)d_in[0];
  const float* coords = (const float*)d_in[1];
  const float* gamma  = (const float*)d_in[2];
  const float* beta   = (const float*)d_in[3];
  const float* w_qkv  = (const float*)d_in[4];
  const float* w_out  = (const float*)d_in[5];

  char* ws = (char*)d_ws;
  size_t off = 0;
  auto take = [&](size_t bytes) {
    void* p = ws + off;
    off = (off + bytes + 255) & ~(size_t)255;
    return p;
  };
  __bf16* xn   = (__bf16*)take((size_t)NTOK * DIM * 2);
  __bf16* wq   = (__bf16*)take((size_t)EQKV * DIM * 2);
  __bf16* wo   = (__bf16*)take((size_t)DIM * DIM * 2);
  __bf16* qkv  = (__bf16*)take((size_t)NTOK * EQKV * 2);
  __bf16* qr   = (__bf16*)take((size_t)NTOK * DIM * 2);
  __bf16* kr   = (__bf16*)take((size_t)NTOK * DIM * 2);
  __bf16* vt   = (__bf16*)take((size_t)NTOK * DIM * 2);
  __bf16* obuf = (__bf16*)take((size_t)NTOK * DIM * 2);

  cast_bf16_kernel<<<(EQKV * DIM + 255) / 256, 256, 0, stream>>>(w_qkv, wq, EQKV * DIM);
  cast_bf16_kernel<<<(DIM * DIM + 255) / 256, 256, 0, stream>>>(w_out, wo, DIM * DIM);
  ln_cast_kernel<<<NTOK, 256, 0, stream>>>(x, gamma, beta, xn);
  gemm_bf16_nt<__bf16><<<dim3(EQKV / 64, NTOK / 512), 256, 0, stream>>>(
      xn, wq, qkv, NTOK, EQKV, DIM);
  rope_split_kernel<<<BATCH * SEQ * HEADS, 96, 0, stream>>>(qkv, coords, qr, kr, vt);
  attn_kernel<<<dim3(BATCH * HEADS, SEQ / 256), 256, 0, stream>>>(qr, kr, vt, obuf);
  gemm_bf16_nt<float><<<dim3(DIM / 64, NTOK / 512), 256, 0, stream>>>(
      obuf, wo, (float*)d_out, NTOK, DIM, DIM);
}
